// DeepEdgeConvolution_72301479461283
// MI455X (gfx1250) — matrix operand order repaired and checked by reading, not compiled, not run on hardware
//
#include <hip/hip_runtime.h>

// CDNA5 / gfx1250 (MI455X) fused edge-MLP + scatter-reduce.
// All matmuls via v_wmma_f32_16x16x32_f16; BN folded into following weights.

typedef __attribute__((ext_vector_type(16))) _Float16 v16h;
typedef __attribute__((ext_vector_type(8)))  _Float16 v8h;
typedef __attribute__((ext_vector_type(8)))  float    v8f;

static constexpr int Dc = 64;     // edge feature dim
static constexpr int Hc = 128;    // hidden dim (and output K)
static constexpr int Kc = 128;    // output channels
static constexpr int Bc = 32;     // batch
static constexpr int TILE_M = 128;
static constexpr float BN_EPS = 1e-5f;

// ---------------------------------------------------------------------------
// WMMA core: one wave computes rows [rowBase .. rowBase+15] x all 128 columns.
// A fragment layout (16-bit A 16x32, contiguous k; B packed to match):
//   lane<16 : v0..3 = K 0..7,  v4..7 = K 8..15   (row = lane)
//   lane>=16: v0..3 = K 16..23, v4..7 = K 24..31   (row = lane-16)
// B fragments are pre-packed so lane L reads 8 contiguous dwords.
// ---------------------------------------------------------------------------
template <int KDIM, bool IN_F32>
__device__ inline void wmma_rows(const void* Xv, size_t rowBase,
                                 const v16h* __restrict__ Wp,
                                 int lane, v8f acc[8]) {
  constexpr int KSTEPS = KDIM / 32;
  const int rrow = lane & 15;
  const int koff = ((lane >> 4) & 1) * 16;
#pragma unroll
  for (int ks = 0; ks < KSTEPS; ++ks) {
    v16h a;
    if (IN_F32) {
      const float* ap = (const float*)Xv + (rowBase + rrow) * KDIM + ks * 32 + koff;
#pragma unroll
      for (int i = 0; i < 8; ++i) {
        a[i]     = (_Float16)ap[i];
        a[8 + i] = (_Float16)ap[8 + i];
      }
    } else {
      const _Float16* ap = (const _Float16*)Xv + (rowBase + rrow) * KDIM + ks * 32 + koff;
      v8h lo = *(const v8h*)ap;
      v8h hi = *(const v8h*)(ap + 8);
      a = __builtin_shufflevector(lo, hi, 0,1,2,3,4,5,6,7,8,9,10,11,12,13,14,15);
    }
#pragma unroll
    for (int n = 0; n < 8; ++n) {
      v16h b = Wp[(n * KSTEPS + ks) * 32 + lane];
      acc[n] = __builtin_amdgcn_wmma_f32_16x16x32_f16(
          false, a, false, b, (short)0, acc[n], false, false);
    }
  }
}

// ---------------------------------------------------------------------------
// prep: zero accumulators, pack W0 (f32 [64x128]) into f16 B-fragment layout.
// Packed element (k,n): kt=k/32, kl=k%32, lane=(n&15)+16*(kl/16),
//   dword i=(kl&15)/2, half = kl&1, base dword = ((nt*KSTEPS+kt)*32+lane)*8.
// ---------------------------------------------------------------------------
__global__ void prep_kernel(const float* __restrict__ W0, const float* __restrict__ b0,
                            v16h* W0p, float* b0c,
                            float* stats0, float* stats1,
                            float* accOut, float* rowsum) {
  int t = threadIdx.x;  // 256 threads
  for (int i = t; i < 2 * Hc; i += 256) { stats0[i] = 0.f; stats1[i] = 0.f; }
  for (int i = t; i < Bc * Kc; i += 256) accOut[i] = 0.f;
  if (t < Bc) rowsum[t] = 0.f;
  if (t < Hc) {
    b0c[t] = b0[t];
    const int n = t;
    _Float16* hp = (_Float16*)W0p;
    for (int k = 0; k < Dc; ++k) {
      float w = W0[k * Hc + n];
      int kt = k >> 5, kl = k & 31;
      int lane = (n & 15) + ((kl >> 4) << 4);
      int nt = n >> 4;
      int i = (kl & 15) >> 1;
      int dw = ((nt * (Dc / 32) + kt) * 32 + lane) * 8 + i;
      hp[dw * 2 + (kl & 1)] = (_Float16)w;
    }
  }
}

// ---------------------------------------------------------------------------
// GEMM + ReLU + BN statistics. Writes f16 activations; accumulates per-channel
// sum/sumsq via LDS ds_add_f32 then one global atomic per channel per block.
// C/D layout (ISA): VGPR r: lanes 0-15 -> M=r, lanes 16-31 -> M=8+r; N=lane&15.
// ---------------------------------------------------------------------------
template <bool IN_F32, int KDIM>
__global__ __launch_bounds__(256) void gemm_stats_kernel(
    const void* __restrict__ Xv, const v16h* __restrict__ Wp,
    const float* __restrict__ bias, _Float16* __restrict__ Hout,
    float* __restrict__ statsG) {
  __shared__ float statsS[2 * Hc];
  const int t = threadIdx.x;
  if (t < 2 * Hc) statsS[t] = 0.f;
  __syncthreads();

  const int wave = t >> 5, lane = t & 31;
  const size_t e0 = (size_t)blockIdx.x * TILE_M;
  v8f acc[8] = {};
  wmma_rows<KDIM, IN_F32>(Xv, e0 + wave * 16, Wp, lane, acc);

  const int colbase = lane & 15;
  const int rowadd = (lane >> 4) << 3;
#pragma unroll
  for (int n = 0; n < 8; ++n) {
    const int ch = n * 16 + colbase;
    const float bv = bias[ch];
    float s = 0.f, s2 = 0.f;
#pragma unroll
    for (int r = 0; r < 8; ++r) {
      float v = acc[n][r] + bv;
      v = v > 0.f ? v : 0.f;
      s += v;
      s2 += v * v;
      const size_t row = e0 + wave * 16 + rowadd + r;
      Hout[row * Hc + ch] = (_Float16)v;
    }
    atomicAdd(&statsS[ch], s);        // ds_add_f32
    atomicAdd(&statsS[Hc + ch], s2);  // ds_add_f32
  }
  __syncthreads();
  if (t < 2 * Hc) atomicAdd(&statsG[t], statsS[t]);
}

// ---------------------------------------------------------------------------
// Finalize BN stats and fold scale/shift into the next layer:
//   W'[k][n] = scale[k]*W[k][n]  (packed f16 B-layout)
//   b'[n]    = b[n] + sum_k shift[k]*W[k][n]
// One block, 128 threads (thread = output column n).
// ---------------------------------------------------------------------------
__global__ void fold_kernel(const float* __restrict__ statsG,
                            const float* __restrict__ g, const float* __restrict__ bt,
                            const float* __restrict__ Wnext, const float* __restrict__ bnext,
                            v16h* Wp, float* bc, float Einv) {
  __shared__ float scaleS[Hc], shiftS[Hc];
  const int t = threadIdx.x;  // 128
  {
    float mu = statsG[t] * Einv;
    float var = statsG[Hc + t] * Einv - mu * mu;
    float sc = g[t] * rsqrtf(var + BN_EPS);
    scaleS[t] = sc;
    shiftS[t] = bt[t] - mu * sc;
  }
  __syncthreads();
  const int n = t;
  float bacc = bnext[n];
  _Float16* hp = (_Float16*)Wp;
  for (int k = 0; k < Hc; ++k) {
    float w = Wnext[k * Hc + n];
    bacc += shiftS[k] * w;
    float wf = scaleS[k] * w;
    int kt = k >> 5, kl = k & 31;
    int lane = (n & 15) + ((kl >> 4) << 4);
    int nt = n >> 4;
    int i = (kl & 15) >> 1;
    int dw = ((nt * (Hc / 32) + kt) * 32 + lane) * 8 + i;
    hp[dw * 2 + (kl & 1)] = (_Float16)wf;
  }
  bc[n] = bacc;
}

// ---------------------------------------------------------------------------
// Layer-2 GEMM fused with bei gather and the [32,E]x[E,128] reduction.
// h2 tile stays in LDS (f16); bei tile built by gathering edge_nodes at
// src/dst (6.4 MB table -> L2 resident). Output accumulated with
// global_atomic_add_f32 into a 32x128 buffer + 32 row sums.
// ---------------------------------------------------------------------------
__global__ __launch_bounds__(256) void gemm_fused_kernel(
    const _Float16* __restrict__ Hin, const v16h* __restrict__ W2p,
    const float* __restrict__ b2c, const float* __restrict__ edge_nodes,
    const int* __restrict__ src, const int* __restrict__ dst,
    float* __restrict__ accOut, float* __restrict__ rowsumG, int Nnodes) {
  __shared__ _Float16 h2s[TILE_M * Kc];   // 32 KB
  __shared__ float beis[Bc * TILE_M];     // 16 KB
  __shared__ int srcS[TILE_M], dstS[TILE_M];

  const int t = threadIdx.x;
  const size_t e0 = (size_t)blockIdx.x * TILE_M;
  if (t < TILE_M) srcS[t] = src[e0 + t];
  else            dstS[t - TILE_M] = dst[e0 + t - TILE_M];
  __syncthreads();

  const int wave = t >> 5, lane = t & 31;
  v8f acc[8] = {};
  wmma_rows<Hc, false>(Hin, e0 + wave * 16, W2p, lane, acc);

  const int colbase = lane & 15;
  const int rowadd = (lane >> 4) << 3;
#pragma unroll
  for (int n = 0; n < 8; ++n) {
    const int ch = n * 16 + colbase;
    const float bv = b2c[ch];
#pragma unroll
    for (int r = 0; r < 8; ++r) {
      const int row = wave * 16 + rowadd + r;
      h2s[row * Kc + ch] = (_Float16)(acc[n][r] + bv);
    }
  }
  // build bei tile: bei[b][e] = edge_nodes[b, src[e]] + edge_nodes[b, dst[e]]
  for (int idx = t; idx < Bc * TILE_M; idx += 256) {
    const int b = idx >> 7, el = idx & (TILE_M - 1);
    beis[idx] = edge_nodes[(size_t)b * Nnodes + srcS[el]] +
                edge_nodes[(size_t)b * Nnodes + dstS[el]];
  }
  __syncthreads();

  if (t < Bc) {
    float s = 0.f;
    for (int e = 0; e < TILE_M; ++e) s += beis[t * TILE_M + e];
    atomicAdd(&rowsumG[t], s);
  }
  // out[b][kb..kb+15] += sum_e bei[b][e] * h2[e][k]
  const int b = t >> 3;
  const int kb = (t & 7) * 16;
  float a16[16];
#pragma unroll
  for (int i = 0; i < 16; ++i) a16[i] = 0.f;
  for (int e = 0; e < TILE_M; ++e) {
    const float gv = beis[b * TILE_M + e];
    const v8h* hp = (const v8h*)(h2s + e * Kc + kb);
    const v8h h0 = hp[0], h1 = hp[1];
#pragma unroll
    for (int i = 0; i < 8; ++i) {
      a16[i]     += gv * (float)h0[i];
      a16[8 + i] += gv * (float)h1[i];
    }
  }
#pragma unroll
  for (int i = 0; i < 16; ++i) atomicAdd(&accOut[b * Kc + kb + i], a16[i]);
}

__global__ void finalize_kernel(const float* __restrict__ accOut,
                                const float* __restrict__ rowsum,
                                float* __restrict__ out) {
  const int t = blockIdx.x * blockDim.x + threadIdx.x;
  if (t >= Bc * Kc) return;
  const float rs = rowsum[t >> 7];
  const float inv = rs > 0.f ? 1.f / rs : 0.f;
  out[t] = accOut[t] * inv;
}

// ---------------------------------------------------------------------------
extern "C" void kernel_launch(void* const* d_in, const int* in_sizes, int n_in,
                              void* d_out, int out_size, void* d_ws, size_t ws_size,
                              hipStream_t stream) {
  const float* edge_nodes = (const float*)d_in[0];
  const float* edge_feats = (const float*)d_in[1];
  const int*   src        = (const int*)d_in[2];
  const int*   dst        = (const int*)d_in[3];
  const float* W0  = (const float*)d_in[4];
  const float* b0  = (const float*)d_in[5];
  const float* g0  = (const float*)d_in[6];
  const float* bt0 = (const float*)d_in[7];
  const float* W1  = (const float*)d_in[8];
  const float* b1  = (const float*)d_in[9];
  const float* g1  = (const float*)d_in[10];
  const float* bt1 = (const float*)d_in[11];
  const float* W2  = (const float*)d_in[12];
  const float* b2  = (const float*)d_in[13];
  float* out = (float*)d_out;

  const int E = in_sizes[2];
  const int Nnodes = in_sizes[0] / Bc;

  char* p = (char*)d_ws;
  auto carve = [&](size_t bytes) {
    void* r = (void*)p;
    p += (bytes + 255) & ~(size_t)255;
    return r;
  };
  v16h* W0p   = (v16h*)carve((size_t)8 * (Dc / 32) * 32 * 32);  // 16 KB
  v16h* W1p   = (v16h*)carve((size_t)8 * (Hc / 32) * 32 * 32);  // 32 KB
  v16h* W2p   = (v16h*)carve((size_t)8 * (Hc / 32) * 32 * 32);  // 32 KB
  float* b0c  = (float*)carve(Hc * 4);
  float* b1c  = (float*)carve(Hc * 4);
  float* b2c  = (float*)carve(Kc * 4);
  float* st0  = (float*)carve(2 * Hc * 4);
  float* st1  = (float*)carve(2 * Hc * 4);
  float* accO = (float*)carve((size_t)Bc * Kc * 4);
  float* rsum = (float*)carve(Bc * 4);
  _Float16* hbuf = (_Float16*)carve((size_t)E * Hc * 2);        // 205 MB

  const int nblk = E / TILE_M;  // E = 800000 -> 6250
  const float Einv = 1.0f / (float)E;

  prep_kernel<<<1, 256, 0, stream>>>(W0, b0, W0p, b0c, st0, st1, accO, rsum);
  gemm_stats_kernel<true, Dc><<<nblk, 256, 0, stream>>>(edge_feats, W0p, b0c, hbuf, st0);
  fold_kernel<<<1, 128, 0, stream>>>(st0, g0, bt0, W1, b1, W1p, b1c, Einv);
  gemm_stats_kernel<false, Hc><<<nblk, 256, 0, stream>>>(hbuf, W1p, b1c, hbuf, st1);
  fold_kernel<<<1, 128, 0, stream>>>(st1, g1, bt1, W2, b2, W2p, b2c, Einv);
  gemm_fused_kernel<<<nblk, 256, 0, stream>>>(hbuf, W2p, b2c, edge_nodes, src, dst,
                                              accO, rsum, Nnodes);
  finalize_kernel<<<(Bc * Kc + 255) / 256, 256, 0, stream>>>(accO, rsum, out);
}
